// ComplexMultiheadAttention_25391846654007
// MI455X (gfx1250) — matrix operand order repaired and checked by reading, not compiled, not run on hardware
//
#include <hip/hip_runtime.h>

// ---------------- problem constants ----------------
#define BB    2
#define NN    2048
#define DIMC  256
#define HH    8
#define DHD   32
#define BN    (BB*NN)       // 4096 rows of (b,n)
#define KAUG  (2*DIMC)      // 512  augmented K for complex-as-real GEMM
#define PCOLS (6*DIMC)      // 1536 cols: [qr|qi|kr|ki|vr|vi]

typedef __attribute__((ext_vector_type(16))) __bf16 v16bf;
typedef __attribute__((ext_vector_type(8)))  float  v8f;

#define WMMA_BF16(a,b,c) \
  __builtin_amdgcn_wmma_f32_16x16x32_bf16(false,(a),false,(b),(short)0,(c),false,false)

// ---- gfx1250 async global->LDS path (guarded: falls back cleanly) --------
#if defined(__gfx1250__) && __has_builtin(__builtin_amdgcn_global_load_async_to_lds_b128)
#define USE_ASYNC_LDS 1
#else
#define USE_ASYNC_LDS 0
#endif

#if USE_ASYNC_LDS
// builtin expects pointers to 16-byte int vectors in AS1 (global) / AS3 (LDS)
typedef int vs4i __attribute__((vector_size(16)));
typedef __attribute__((address_space(1))) vs4i g_vs4i;
typedef __attribute__((address_space(3))) vs4i l_vs4i;
static __device__ inline void async_copy32B(const __bf16* g, __bf16* l) {
  // INST_OFFSET is added to both global and LDS addresses, so one pointer
  // pair covers both 16B halves.
  __builtin_amdgcn_global_load_async_to_lds_b128((g_vs4i*)g, (l_vs4i*)l, 0, 0);
  __builtin_amdgcn_global_load_async_to_lds_b128((g_vs4i*)g, (l_vs4i*)l, 16, 0);
}
static __device__ inline void wait_async0() {
#if __has_builtin(__builtin_amdgcn_s_wait_asynccnt)
  __builtin_amdgcn_s_wait_asynccnt(0);
#else
  asm volatile("s_wait_asynccnt 0" ::: "memory");
#endif
}
#endif

// ---- cross-lane xor reductions within 16-lane rows -------------------------
// Fused DPP16 row_xmask reduction ops: one VALU instruction per step
// (vs mov_dpp + canonicalize + max when the compiler lowers shuffles).
#if defined(__gfx1250__) && __has_builtin(__builtin_amdgcn_update_dpp)
template <int MASK>
static __device__ inline float xor_bcast(float x) {
  return __int_as_float(__builtin_amdgcn_update_dpp(
      0, __float_as_int(x), 0x160 | MASK, 0xf, 0xf, true));
}
#else
template <int MASK>
static __device__ inline float xor_bcast(float x) { return __shfl_xor(x, MASK); }
#endif

#if defined(__gfx1250__)
#define ROWRED_MAX(r, mask)                                                    \
  do {                                                                         \
    float _d;                                                                  \
    asm("v_max_num_f32_dpp %0, %1, %2 row_xmask:" #mask                        \
        " row_mask:0xf bank_mask:0xf bound_ctrl:1"                             \
        : "=v"(_d) : "v"(r), "v"(r));                                          \
    (r) = _d;                                                                  \
  } while (0)
#define ROWRED_ADD(r, mask)                                                    \
  do {                                                                         \
    float _d;                                                                  \
    asm("v_add_f32_dpp %0, %1, %2 row_xmask:" #mask                            \
        " row_mask:0xf bank_mask:0xf bound_ctrl:1"                             \
        : "=v"(_d) : "v"(r), "v"(r));                                          \
    (r) = _d;                                                                  \
  } while (0)
#else
#define ROWRED_MAX(r, mask) (r) = fmaxf((r), xor_bcast<mask>(r))
#define ROWRED_ADD(r, mask) (r) = (r) + xor_bcast<mask>(r)
#endif

// Load a 16-element bf16 A/B fragment from two 16-byte aligned chunks.
static __device__ inline v16bf load_pair16(const __bf16* p0, const __bf16* p1) {
  union { uint4 u[2]; v16bf v; } t;
  t.u[0] = *(const uint4*)p0;
  t.u[1] = *(const uint4*)p1;
  return t.v;
}
static __device__ inline v16bf load_contig16(const __bf16* p) {
  return load_pair16(p, p + 8);
}

// ---------------- prep kernels (bf16 packing of augmented operands) -------

// Xaug [BN x 512] = [x_real | x_imag] in bf16
__global__ __launch_bounds__(256) void prep_x_kernel(
    const float* __restrict__ xr, const float* __restrict__ xi,
    __bf16* __restrict__ Xaug) {
  int i = blockIdx.x * 256 + threadIdx.x;
  if (i >= BN * KAUG) return;
  int row = i >> 9, j = i & 511;
  float v = (j < DIMC) ? xr[row * DIMC + j] : xi[row * DIMC + (j - DIMC)];
  Xaug[i] = (__bf16)v;
}

// Wproj [512 x 1536]: columns grouped [qr|qi|kr|ki|vr|vi]; top half multiplies
// x_real, bottom half multiplies x_imag (with sign for the real parts).
__global__ __launch_bounds__(256) void prep_wproj_kernel(
    const float* __restrict__ wq_r, const float* __restrict__ wq_i,
    const float* __restrict__ wkv_r, const float* __restrict__ wkv_i,
    __bf16* __restrict__ Wp) {
  int i = blockIdx.x * 256 + threadIdx.x;
  if (i >= KAUG * PCOLS) return;
  int k = i / PCOLS, j = i % PCOLS;
  int g = j >> 8, c = j & 255;
  int bot = k >> 8, kk = k & 255;
  float v;
  switch (g) {
    case 0:  v = bot ? -wq_i [kk*256 + c]        : wq_r [kk*256 + c];        break;
    case 1:  v = bot ?  wq_r [kk*256 + c]        : wq_i [kk*256 + c];        break;
    case 2:  v = bot ? -wkv_i[kk*512 + c]        : wkv_r[kk*512 + c];        break;
    case 3:  v = bot ?  wkv_r[kk*512 + c]        : wkv_i[kk*512 + c];        break;
    case 4:  v = bot ? -wkv_i[kk*512 + 256 + c]  : wkv_r[kk*512 + 256 + c];  break;
    default: v = bot ?  wkv_r[kk*512 + 256 + c]  : wkv_i[kk*512 + 256 + c];  break;
  }
  Wp[i] = (__bf16)v;
}

// Wout [512 x 512]: cols [out_r | out_i]
__global__ __launch_bounds__(256) void prep_wout_kernel(
    const float* __restrict__ wo_r, const float* __restrict__ wo_i,
    __bf16* __restrict__ Wo) {
  int i = blockIdx.x * 256 + threadIdx.x;
  if (i >= KAUG * KAUG) return;
  int k = i >> 9, j = i & 511;
  int g = j >> 8, c = j & 255;
  int bot = k >> 8, kk = k & 255;
  float v;
  if (g == 0) v = bot ? -wo_i[kk*256 + c] : wo_r[kk*256 + c];
  else        v = bot ?  wo_r[kk*256 + c] : wo_i[kk*256 + c];
  Wo[i] = (__bf16)v;
}

// Transpose V (real & imag) to [b,h,d,n] so attention B-fragments are
// contiguous 32B loads over the key dimension.
__global__ __launch_bounds__(256) void repack_v_kernel(
    const __bf16* __restrict__ Cp, __bf16* __restrict__ Vrt,
    __bf16* __restrict__ Vit) {
  int i = blockIdx.x * 256 + threadIdx.x;   // i = ((b*8+h)*32+d)*2048+n
  if (i >= BB * HH * DHD * NN) return;
  int n = i & 2047, d = (i >> 11) & 31, h = (i >> 16) & 7, b = i >> 19;
  size_t srow = (size_t)(b * NN + n) * PCOLS;
  Vrt[i] = Cp[srow + 4*DIMC + h*DHD + d];
  Vit[i] = Cp[srow + 5*DIMC + h*DHD + d];
}

// ---------------- bf16 WMMA GEMM: C[M x Ncols] = A[M x K] * B[K x Ncols] ---
// MODE 0: store bf16 row-major (ldc = Ncols)
// MODE 1: store f32 interleaved complex (tile-uniform real/imag split since
//         column tiles are 64-wide and DIMC is tile-aligned)
template <int MODE>
__global__ __launch_bounds__(128) void gemm_bf16_kernel(
    const __bf16* __restrict__ A, const __bf16* __restrict__ Bm,
    void* __restrict__ Cout, int M, int Ncols, int K) {
  __shared__ __bf16 As[64][40];   // row-major, +8 pad vs bank conflicts
  __shared__ __bf16 Bs[64][40];   // column-major: Bs[col][k]

  const int tid = threadIdx.x;
  const int lane = tid & 31, w = tid >> 5;
  const int hl = lane >> 4, l16 = lane & 15;
  const int wm = (w >> 1) * 32, wn = (w & 1) * 32;
  const int rowBase = blockIdx.y * 64, colBase = blockIdx.x * 64;

  v8f acc[2][2];
  acc[0][0] = {}; acc[0][1] = {}; acc[1][0] = {}; acc[1][1] = {};

  for (int kb = 0; kb < K; kb += 32) {
    // A tile: 64x32, 2 threads per row, 32B each (async DMA to LDS if avail.)
    {
      int r = tid >> 1, seg = (tid & 1) * 16;
      const __bf16* src = A + (size_t)(rowBase + r) * K + kb + seg;
#if USE_ASYNC_LDS
      async_copy32B(src, &As[r][seg]);
#else
      uint4* dst = (uint4*)&As[r][seg];
      dst[0] = ((const uint4*)src)[0];
      dst[1] = ((const uint4*)src)[1];
#endif
    }
    // B tile: 32x64 loaded coalesced row-major, scattered to column-major LDS
    {
      int k = tid >> 2, cs = (tid & 3) * 16;
      union { uint4 u[2]; __bf16 e[16]; } t;
      const uint4* src = (const uint4*)(Bm + (size_t)(kb + k) * Ncols + colBase + cs);
      t.u[0] = src[0];
      t.u[1] = src[1];
      #pragma unroll
      for (int i = 0; i < 16; ++i) Bs[cs + i][k] = t.e[i];
    }
#if USE_ASYNC_LDS
    wait_async0();
#endif
    __syncthreads();

    v16bf af[2], bf[2];
    #pragma unroll
    for (int mi = 0; mi < 2; ++mi) {
      const __bf16* p = &As[wm + mi * 16 + l16][0];
      af[mi] = load_contig16(p + hl * 16);   // A 16x32 layout
    }
    #pragma unroll
    for (int ni = 0; ni < 2; ++ni) {
      const __bf16* p = &Bs[wn + ni * 16 + l16][hl * 16];  // B 32x16 layout
      bf[ni] = load_contig16(p);
    }
    #pragma unroll
    for (int mi = 0; mi < 2; ++mi)
      #pragma unroll
      for (int ni = 0; ni < 2; ++ni)
        acc[mi][ni] = WMMA_BF16(af[mi], bf[ni], acc[mi][ni]);
    __syncthreads();
  }

  if (MODE == 0) {
    __bf16* o = (__bf16*)Cout;
    #pragma unroll
    for (int mi = 0; mi < 2; ++mi)
      #pragma unroll
      for (int ni = 0; ni < 2; ++ni)
        #pragma unroll
        for (int v = 0; v < 8; ++v) {
          int row = rowBase + wm + mi * 16 + v + hl * 8;  // C frag: M = v+8*half
          int col = colBase + wn + ni * 16 + l16;         //         N = lane%16
          o[(size_t)row * Ncols + col] = (__bf16)acc[mi][ni][v];
        }
  } else {
    // block-uniform real/imag split: whole 64-col tile is on one side of 256
    float* o = (float*)Cout;
    const int imag = (colBase >= DIMC) ? 1 : 0;
    const int cadj = colBase - imag * DIMC;
    #pragma unroll
    for (int mi = 0; mi < 2; ++mi)
      #pragma unroll
      for (int ni = 0; ni < 2; ++ni)
        #pragma unroll
        for (int v = 0; v < 8; ++v) {
          int row = rowBase + wm + mi * 16 + v + hl * 8;
          int col = cadj + wn + ni * 16 + l16;
          o[((size_t)row * DIMC + col) * 2 + imag] = acc[mi][ni][v];
        }
  }
}

// ---------------- flash attention over 4 complex replicas ------------------
// grid: x = b*H+h (16), y = query tile of 64 (32). 4 waves, 16 queries/wave.
__global__ __launch_bounds__(128) void cmha_attn_kernel(
    const __bf16* __restrict__ Cp, const __bf16* __restrict__ Vrt,
    const __bf16* __restrict__ Vit, __bf16* __restrict__ Oaug) {
  __shared__ __bf16 Ps[4][16 * 32];   // per-wave P relayout scratch

  const int tid = threadIdx.x;
  const int lane = tid & 31, w = tid >> 5;
  const int hl = lane >> 4, l16 = lane & 15;
  const int bh = blockIdx.x, b = bh >> 3, h = bh & 7;
  const int qbase = blockIdx.y * 64 + w * 16;
  const float scale = 0.17677669529663687f;   // 1/sqrt(DH)
  const size_t vbase = (size_t)(b * HH + h) * DHD * NN;
  __bf16* Pl = &Ps[w][0];

  v8f or0 = {}, or1 = {}, oi0 = {}, oi1 = {};

  for (int t = 0; t < 4; ++t) {
    const int qc = ((t >= 2) ? DIMC : 0) + h * DHD;           // qr / qi
    const int kc = ((t & 1) ? 3 * DIMC : 2 * DIMC) + h * DHD; // kr / ki
    // sign-recombination coefficients
    const float c_rr = (t == 0) ? 1.f : (t == 3) ? -1.f : 0.f;
    const float c_ri = (t == 1 || t == 2) ? -1.f : 0.f;
    const float c_ir = (t == 1 || t == 2) ? 1.f : 0.f;
    const float c_ii = (t == 0) ? 1.f : (t == 3) ? -1.f : 0.f;

    const __bf16* qp = Cp + (size_t)(b * NN + qbase + l16) * PCOLS + qc;
    v16bf qf = load_contig16(qp + hl * 16);
    // fold 1/sqrt(DH) into Q once per replica instead of per key tile
    #pragma unroll
    for (int e = 0; e < 16; ++e) qf[e] = (__bf16)((float)qf[e] * scale);

    v8f m, l = {}, ar0 = {}, ar1 = {}, ai0 = {}, ai1 = {};
    #pragma unroll
    for (int v = 0; v < 8; ++v) m[v] = -1e30f;

    for (int kt = 0; kt < NN; kt += 32) {
      const __bf16* kp0 =
          Cp + (size_t)(b * NN + kt + l16) * PCOLS + kc + hl * 16;
      const __bf16* vr0 = Vrt + vbase + (size_t)l16 * NN + kt + hl * 16;
      const __bf16* vi0 = Vit + vbase + (size_t)l16 * NN + kt + hl * 16;
      if (kt + 32 < NN) {   // uniform branch: prefetch next tiles
        __builtin_prefetch(kp0 + (size_t)32 * PCOLS, 0, 3);
        __builtin_prefetch(vr0 + 32, 0, 3);
        __builtin_prefetch(vi0 + 32, 0, 3);
      }
      v16bf kf0 = load_contig16(kp0);
      v16bf kf1 = load_contig16(kp0 + (size_t)16 * PCOLS);
      v8f z = {};
      v8f s0 = WMMA_BF16(qf, kf0, z);
      v8f s1 = WMMA_BF16(qf, kf1, z);

      v8f mn, corr;
      #pragma unroll
      for (int v = 0; v < 8; ++v) {
        float a = s0[v], c = s1[v];
        float r = fmaxf(a, c);
        ROWRED_MAX(r, 1);
        ROWRED_MAX(r, 2);
        ROWRED_MAX(r, 4);
        ROWRED_MAX(r, 8);
        mn[v] = fmaxf(m[v], r);
        corr[v] = __expf(m[v] - mn[v]);
        float p0 = __expf(a - mn[v]);
        float p1 = __expf(c - mn[v]);
        s0[v] = p0; s1[v] = p1;
        float rs = p0 + p1;
        ROWRED_ADD(rs, 1);
        ROWRED_ADD(rs, 2);
        ROWRED_ADD(rs, 4);
        ROWRED_ADD(rs, 8);
        l[v] = l[v] * corr[v] + rs;
        m[v] = mn[v];
      }
      #pragma unroll
      for (int v = 0; v < 8; ++v) {
        ar0[v] *= corr[v]; ar1[v] *= corr[v];
        ai0[v] *= corr[v]; ai1[v] *= corr[v];
      }
      // relayout P: C-frag -> A-frag through per-wave LDS
      #pragma unroll
      for (int v = 0; v < 8; ++v) {
        int q = v + hl * 8;
        Pl[q * 32 + l16]      = (__bf16)s0[v];
        Pl[q * 32 + 16 + l16] = (__bf16)s1[v];
      }
      const __bf16* pp = Pl + l16 * 32;
      v16bf pf = load_contig16(pp + hl * 16);

      ar0 = WMMA_BF16(pf, load_contig16(vr0), ar0);
      ar1 = WMMA_BF16(pf, load_contig16(vr0 + (size_t)16 * NN), ar1);
      ai0 = WMMA_BF16(pf, load_contig16(vi0), ai0);
      ai1 = WMMA_BF16(pf, load_contig16(vi0 + (size_t)16 * NN), ai1);
    }

    #pragma unroll
    for (int v = 0; v < 8; ++v) {
      float inv = 1.0f / l[v];
      or0[v] += c_rr * (ar0[v] * inv) + c_ri * (ai0[v] * inv);
      or1[v] += c_rr * (ar1[v] * inv) + c_ri * (ai1[v] * inv);
      oi0[v] += c_ir * (ar0[v] * inv) + c_ii * (ai0[v] * inv);
      oi1[v] += c_ir * (ar1[v] * inv) + c_ii * (ai1[v] * inv);
    }
  }

  // store merged heads: Oaug [BN x 512] = [o_r | o_i]
  const int colr = h * DHD + l16;
  #pragma unroll
  for (int v = 0; v < 8; ++v) {
    size_t row = (size_t)(b * NN + qbase + v + hl * 8);
    Oaug[row * KAUG + colr]              = (__bf16)or0[v];
    Oaug[row * KAUG + colr + 16]         = (__bf16)or1[v];
    Oaug[row * KAUG + DIMC + colr]       = (__bf16)oi0[v];
    Oaug[row * KAUG + DIMC + colr + 16]  = (__bf16)oi1[v];
  }
}

// ---------------- launch ---------------------------------------------------
extern "C" void kernel_launch(void* const* d_in, const int* in_sizes, int n_in,
                              void* d_out, int out_size, void* d_ws,
                              size_t ws_size, hipStream_t stream) {
  const float* x_real = (const float*)d_in[0];
  const float* x_imag = (const float*)d_in[1];
  const float* wq_r   = (const float*)d_in[2];
  const float* wq_i   = (const float*)d_in[3];
  const float* wkv_r  = (const float*)d_in[4];
  const float* wkv_i  = (const float*)d_in[5];
  const float* wo_r   = (const float*)d_in[6];
  const float* wo_i   = (const float*)d_in[7];

  char* ws = (char*)d_ws;
  __bf16* Xaug  = (__bf16*)(ws);                       // 4096*512  = 4 MB
  __bf16* Wproj = (__bf16*)(ws + 4194304);             // 512*1536  = 1.5 MB
  __bf16* Wout  = (__bf16*)(ws + 5767168);             // 512*512   = 0.5 MB
  __bf16* Cproj = (__bf16*)(ws + 6291456);             // 4096*1536 = 12 MB
  __bf16* Vrt   = (__bf16*)(ws + 18874368);            // 2*8*32*2048 = 2 MB
  __bf16* Vit   = (__bf16*)(ws + 20971520);            // 2 MB
  __bf16* Oaug  = (__bf16*)(ws + 23068672);            // 4096*512  = 4 MB

  prep_x_kernel<<<(BN * KAUG) / 256, 256, 0, stream>>>(x_real, x_imag, Xaug);
  prep_wproj_kernel<<<(KAUG * PCOLS) / 256, 256, 0, stream>>>(wq_r, wq_i,
                                                              wkv_r, wkv_i, Wproj);
  prep_wout_kernel<<<(KAUG * KAUG) / 256, 256, 0, stream>>>(wo_r, wo_i, Wout);

  // fused complex QKV projection: [xr|xi] @ Wproj -> [qr|qi|kr|ki|vr|vi] bf16
  gemm_bf16_kernel<0><<<dim3(PCOLS / 64, BN / 64), 128, 0, stream>>>(
      Xaug, Wproj, Cproj, BN, PCOLS, KAUG);

  repack_v_kernel<<<(BB * HH * DHD * NN) / 256, 256, 0, stream>>>(Cproj, Vrt, Vit);

  cmha_attn_kernel<<<dim3(BB * HH, NN / 64), 128, 0, stream>>>(Cproj, Vrt, Vit,
                                                               Oaug);

  // fused complex output projection, f32 interleaved (b,n,d,2) store
  gemm_bf16_kernel<1><<<dim3(KAUG / 64, BN / 64), 128, 0, stream>>>(
      Oaug, Wout, d_out, BN, KAUG, KAUG);
}
